// RelationNonLocal_77867757077173
// MI455X (gfx1250) — matrix operand order repaired and checked by reading, not compiled, not run on hardware
//
#include <hip/hip_runtime.h>

#define C_DIM  256
#define HW_DIM 2304
#define N_BATCH 16
#define STRIP   96      // rows per block strip (2304 = 24 * 96)
#define LDA    264      // padded bf16 row stride in LDS

typedef __attribute__((ext_vector_type(16))) __bf16       v16bf;
typedef __attribute__((ext_vector_type(8)))  float        v8f;
typedef __attribute__((ext_vector_type(4)))  unsigned int u32x4;
typedef __attribute__((ext_vector_type(2)))  unsigned int u32x2;

union Frag { unsigned int u[8]; v16bf v; };

__device__ __forceinline__ unsigned short f2bf(float f) {
  union { __bf16 h; unsigned short u; } p; p.h = (__bf16)f; return p.u;
}
__device__ __forceinline__ unsigned int pack2bf(float lo, float hi) {
  union { __bf16 h[2]; unsigned int u; } p;
  p.h[0] = (__bf16)lo; p.h[1] = (__bf16)hi; return p.u;
}
// A-matrix (16x32 bf16) K index for VGPR slot v, lane-half h
__device__ __forceinline__ int pairKA(int v, int h) {
  return ((v & 3) << 1) + ((v >> 2) << 4) + (h << 3);
}
// B-matrix (32x16 bf16) K index for VGPR slot v, lane-half h
__device__ __forceinline__ int pairKB(int v, int h) {
  return ((v & 3) << 1) + ((v >> 2) << 4) + (h << 3);
}
__device__ __forceinline__ v8f wmma_bf16(Frag a, Frag b, v8f c) {
  return __builtin_amdgcn_wmma_f32_16x16x32_bf16(false, a.v, false, b.v,
                                                 (short)0, c, false, false);
}

// ---------------------------------------------------------------------------
// Kernel 0: convert the four 256x256 f32 weights to bf16 once.
// ---------------------------------------------------------------------------
__global__ __launch_bounds__(256) void cvtw_kernel(
    const float* __restrict__ W0, const float* __restrict__ W1,
    const float* __restrict__ W2, const float* __restrict__ W3,
    unsigned short* __restrict__ Wbf)
{
  const int t = blockIdx.x * 256 + threadIdx.x;      // 0..16383, 4 elems each
  const float* Ws[4] = {W0, W1, W2, W3};
  #pragma unroll
  for (int m = 0; m < 4; ++m) {
    float4 v = *(const float4*)(Ws[m] + t * 4);
    u32x2 pk;
    pk[0] = pack2bf(v.x, v.y);
    pk[1] = pack2bf(v.z, v.w);
    *(u32x2*)&Wbf[m * (C_DIM * C_DIM) + t * 4] = pk;
  }
}

// ---------------------------------------------------------------------------
// Kernel 1: projections. X strip staged once in LDS (bf16).
//   fvT, fkT stored TRANSPOSED [C][HW] (computed as f^T = W @ X, X-tile as B)
//   fq  stored normal [HW][C]           (X-tile as A, W as B)
// ---------------------------------------------------------------------------
__global__ __launch_bounds__(192) void proj3_kernel(
    const float* __restrict__ x, const unsigned short* __restrict__ Wbf,
    const float* __restrict__ b0, const float* __restrict__ b1,
    const float* __restrict__ b2,
    unsigned short* __restrict__ fvT, unsigned short* __restrict__ fkT,
    unsigned short* __restrict__ fq)
{
  __shared__ unsigned short ldsA[STRIP * LDA];       // 96 x 264 bf16
  const int n    = blockIdx.x;
  const int i0   = blockIdx.y * STRIP;
  const int tid  = threadIdx.x;
  const int lane = tid & 31, wave = tid >> 5;
  const int mrow = lane & 15, half = lane >> 4;

  // stage X^T tile [96 i x 256 c] bf16 ; float4 loads along i
  const float* xb = x + n * (C_DIM * HW_DIM) + i0;
  for (int j = 0; j < 32; ++j) {
    int q  = j * 192 + tid;          // 6144 quads
    int c  = q / 24;
    int i  = (q - c * 24) * 4;
    float4 v = *(const float4*)(xb + c * HW_DIM + i);
    unsigned short* d = &ldsA[i * LDA + c];
    d[0]       = f2bf(v.x);
    d[LDA]     = f2bf(v.y);
    d[2 * LDA] = f2bf(v.z);
    d[3 * LDA] = f2bf(v.w);
  }
  __syncthreads();

  const float* bias01[2] = {b0, b1};
  unsigned short* outT[2] = {fvT, fkT};

  // --- fvT / fkT : transposed orientation (M-dim = o, N-dim = i) ---
  for (int wi = 0; wi < 2; ++wi) {
    const unsigned short* Wb = Wbf + wi * (C_DIM * C_DIM);
    v8f acc[16];
    #pragma unroll
    for (int t = 0; t < 16; ++t) acc[t] = v8f{0,0,0,0,0,0,0,0};
    for (int k0 = 0; k0 < C_DIM; k0 += 32) {
      Frag bf;                                       // B[k=c][n=i] from LDS
      #pragma unroll
      for (int v = 0; v < 8; ++v)
        bf.u[v] = *(const unsigned int*)
            &ldsA[(wave * 16 + mrow) * LDA + k0 + pairKB(v, half)];
      #pragma unroll
      for (int ot = 0; ot < 16; ++ot) {
        Frag a;                                      // A[m=o][k=c] from Wbf
        #pragma unroll
        for (int v = 0; v < 8; ++v)
          a.u[v] = *(const unsigned int*)
              &Wb[(ot * 16 + mrow) * C_DIM + k0 + pairKA(v, half)];
        acc[ot] = wmma_bf16(a, bf, acc[ot]);
      }
    }
    unsigned short* dst = outT[wi] + n * (C_DIM * HW_DIM);
    const float* bb = bias01[wi];
    #pragma unroll
    for (int ot = 0; ot < 16; ++ot)
      #pragma unroll
      for (int v = 0; v < 8; ++v) {
        int o = ot * 16 + v + 8 * half;
        int i = i0 + wave * 16 + mrow;               // lanes -> consecutive i
        dst[o * HW_DIM + i] = f2bf(acc[ot][v] + bb[o]);
      }
  }

  // --- fq : normal orientation (M-dim = i, N-dim = o), layout [HW][C] ---
  {
    const unsigned short* Wb = Wbf + 2 * (C_DIM * C_DIM);
    v8f acc[16];
    #pragma unroll
    for (int t = 0; t < 16; ++t) acc[t] = v8f{0,0,0,0,0,0,0,0};
    for (int k0 = 0; k0 < C_DIM; k0 += 32) {
      Frag a;                                        // A[m=i][k=c] from LDS
      #pragma unroll
      for (int v = 0; v < 8; ++v)
        a.u[v] = *(const unsigned int*)
            &ldsA[(wave * 16 + mrow) * LDA + k0 + pairKA(v, half)];
      #pragma unroll
      for (int ct = 0; ct < 16; ++ct) {
        Frag b;                                      // B[k=c][n=o] from Wbf
        #pragma unroll
        for (int v = 0; v < 8; ++v)
          b.u[v] = *(const unsigned int*)
              &Wb[(ct * 16 + mrow) * C_DIM + k0 + pairKB(v, half)];
        acc[ct] = wmma_bf16(a, b, acc[ct]);
      }
    }
    unsigned short* dst = fq + n * (HW_DIM * C_DIM);
    #pragma unroll
    for (int ct = 0; ct < 16; ++ct) {
      float bias = b2[ct * 16 + mrow];
      #pragma unroll
      for (int v = 0; v < 8; ++v) {
        int row = i0 + wave * 16 + v + 8 * half;
        dst[row * C_DIM + ct * 16 + mrow] = f2bf(acc[ct][v] + bias);
      }
    }
  }
}

// ---------------------------------------------------------------------------
// Kernel 2: Amat[b][a] = (1/HW) * sum_j fk[j][a] * fv[j][b]  (bf16, per batch)
// fkT/fvT are [C][HW] so both A and B fragments are contiguous dword loads.
// No LDS needed. grid (n, cg<4): 256 rows x 64 cols per block.
// ---------------------------------------------------------------------------
__global__ __launch_bounds__(256) void ktv_kernel(
    const unsigned short* __restrict__ fkT, const unsigned short* __restrict__ fvT,
    unsigned short* __restrict__ Amat)
{
  const int n    = blockIdx.x;
  const int cg   = blockIdx.y;                 // 64-col (b) group
  const int tid  = threadIdx.x;
  const int lane = tid & 31, wave = tid >> 5;
  const int mrow = lane & 15, half = lane >> 4;
  const float invHW = 1.0f / (float)HW_DIM;
  const unsigned short* ka = fkT + n * (C_DIM * HW_DIM);
  const unsigned short* vb = fvT + n * (C_DIM * HW_DIM) + (cg * 64) * HW_DIM;

  v8f acc[2][4];
  #pragma unroll
  for (int r = 0; r < 2; ++r)
    #pragma unroll
    for (int c = 0; c < 4; ++c) acc[r][c] = v8f{0,0,0,0,0,0,0,0};

  for (int j0 = 0; j0 < HW_DIM; j0 += 32) {
    Frag a[2], b[4];
    #pragma unroll
    for (int r = 0; r < 2; ++r)
      #pragma unroll
      for (int v = 0; v < 8; ++v)
        a[r].u[v] = *(const unsigned int*)
            &ka[((wave * 2 + r) * 16 + mrow) * HW_DIM + j0 + pairKA(v, half)];
    #pragma unroll
    for (int c = 0; c < 4; ++c)
      #pragma unroll
      for (int v = 0; v < 8; ++v)
        b[c].u[v] = *(const unsigned int*)
            &vb[(c * 16 + mrow) * HW_DIM + j0 + pairKB(v, half)];
    #pragma unroll
    for (int r = 0; r < 2; ++r)
      #pragma unroll
      for (int c = 0; c < 4; ++c)
        acc[r][c] = wmma_bf16(a[r], b[c], acc[r][c]);
  }

  unsigned short* Ab = Amat + n * (C_DIM * C_DIM);
  #pragma unroll
  for (int r = 0; r < 2; ++r)
    #pragma unroll
    for (int c = 0; c < 4; ++c) {
      int a0   = (wave * 2 + r) * 16 + 8 * half;   // 8 consecutive a (rows)
      int bcol = cg * 64 + c * 16 + mrow;
      u32x4 pk;
      #pragma unroll
      for (int j = 0; j < 4; ++j)
        pk[j] = pack2bf(acc[r][c][2 * j] * invHW, acc[r][c][2 * j + 1] * invHW);
      *(u32x4*)&Ab[bcol * C_DIM + a0] = pk;        // one b128 store per tile
    }
}

// ---------------------------------------------------------------------------
// Kernel 3: T^T = Amat @ fq^T (bf16 in LDS), then out^T = W3 @ T^T + b3.
// All fragments are contiguous dword loads; T stored with one b128 per tile.
// Coalesced NCHW f32 output stores.
// ---------------------------------------------------------------------------
__global__ __launch_bounds__(192) void qm_out_kernel(
    const unsigned short* __restrict__ fq, const unsigned short* __restrict__ Amat,
    const unsigned short* __restrict__ W3b, const float* __restrict__ b3,
    float* __restrict__ out)
{
  __shared__ unsigned short Tl[STRIP * LDA];
  const int n    = blockIdx.x;
  const int i0   = blockIdx.y * STRIP;
  const int tid  = threadIdx.x;
  const int lane = tid & 31, wave = tid >> 5;
  const int mrow = lane & 15, half = lane >> 4;
  const unsigned short* fqb = fq + n * (HW_DIM * C_DIM);
  const unsigned short* Ab  = Amat + n * (C_DIM * C_DIM);

  // Phase A: T^T tiles (M-dim = c_out via Amat, N-dim = i via fq)
  {
    v8f acc[16];
    #pragma unroll
    for (int t = 0; t < 16; ++t) acc[t] = v8f{0,0,0,0,0,0,0,0};
    for (int k0 = 0; k0 < C_DIM; k0 += 32) {
      Frag bf;                                     // B[k=c_in][n=i] from fq
      #pragma unroll
      for (int v = 0; v < 8; ++v)
        bf.u[v] = *(const unsigned int*)
            &fqb[(i0 + wave * 16 + mrow) * C_DIM + k0 + pairKB(v, half)];
      #pragma unroll
      for (int ot = 0; ot < 16; ++ot) {
        Frag a;                                    // A[m=c_out][k=c_in]
        #pragma unroll
        for (int v = 0; v < 8; ++v)
          a.u[v] = *(const unsigned int*)
              &Ab[(ot * 16 + mrow) * C_DIM + k0 + pairKA(v, half)];
        acc[ot] = wmma_bf16(a, bf, acc[ot]);
      }
    }
    #pragma unroll
    for (int ot = 0; ot < 16; ++ot) {              // Tl[i][c_out], b128 store
      u32x4 pk;
      #pragma unroll
      for (int j = 0; j < 4; ++j)
        pk[j] = pack2bf(acc[ot][2 * j], acc[ot][2 * j + 1]);
      *(u32x4*)&Tl[(wave * 16 + mrow) * LDA + ot * 16 + 8 * half] = pk;
    }
  }
  __syncthreads();

  // Phase B: out^T = W3 @ T^T + b3
  {
    v8f acc[16];
    #pragma unroll
    for (int t = 0; t < 16; ++t) acc[t] = v8f{0,0,0,0,0,0,0,0};
    for (int k0 = 0; k0 < C_DIM; k0 += 32) {
      Frag bf;                                     // B[k=c][n=i] from Tl
      #pragma unroll
      for (int v = 0; v < 8; ++v)
        bf.u[v] = *(const unsigned int*)
            &Tl[(wave * 16 + mrow) * LDA + k0 + pairKB(v, half)];
      #pragma unroll
      for (int ot = 0; ot < 16; ++ot) {
        Frag a;                                    // A[m=o][k=c] from W3 bf16
        #pragma unroll
        for (int v = 0; v < 8; ++v)
          a.u[v] = *(const unsigned int*)
              &W3b[(ot * 16 + mrow) * C_DIM + k0 + pairKA(v, half)];
        acc[ot] = wmma_bf16(a, bf, acc[ot]);
      }
    }
    float* ob = out + n * (C_DIM * HW_DIM);
    #pragma unroll
    for (int ot = 0; ot < 16; ++ot)
      #pragma unroll
      for (int v = 0; v < 8; ++v) {
        int o = ot * 16 + v + 8 * half;
        int i = i0 + wave * 16 + mrow;             // coalesced f32 stores
        ob[o * HW_DIM + i] = acc[ot][v] + b3[o];
      }
  }
}

extern "C" void kernel_launch(void* const* d_in, const int* in_sizes, int n_in,
                              void* d_out, int out_size, void* d_ws, size_t ws_size,
                              hipStream_t stream) {
  (void)in_sizes; (void)n_in; (void)out_size; (void)ws_size;
  const float* x  = (const float*)d_in[0];
  const float* W0 = (const float*)d_in[1];
  const float* b0 = (const float*)d_in[2];
  const float* W1 = (const float*)d_in[3];
  const float* b1 = (const float*)d_in[4];
  const float* W2 = (const float*)d_in[5];
  const float* b2 = (const float*)d_in[6];
  const float* W3 = (const float*)d_in[7];
  const float* b3 = (const float*)d_in[8];
  float* out = (float*)d_out;

  const size_t FE = (size_t)N_BATCH * HW_DIM * C_DIM;
  const size_t AE = (size_t)N_BATCH * C_DIM * C_DIM;
  unsigned short* fvT  = (unsigned short*)d_ws;    // bf16 [N][C][HW]
  unsigned short* fkT  = fvT + FE;                 // bf16 [N][C][HW]
  unsigned short* fq   = fkT + FE;                 // bf16 [N][HW][C]
  unsigned short* Amat = fq + FE;                  // bf16 [N][C][C] (scaled M^T)
  unsigned short* Wbf  = Amat + AE;                // bf16 4 x [C][C]

  cvtw_kernel<<<dim3(64), 256, 0, stream>>>(W0, W1, W2, W3, Wbf);
  proj3_kernel<<<dim3(N_BATCH, HW_DIM / STRIP), 192, 0, stream>>>(
      x, Wbf, b0, b1, b2, fvT, fkT, fq);
  ktv_kernel<<<dim3(N_BATCH, 4), 256, 0, stream>>>(fkT, fvT, Amat);
  qm_out_kernel<<<dim3(N_BATCH, HW_DIM / STRIP), 192, 0, stream>>>(
      fq, Amat, Wbf + 3 * (C_DIM * C_DIM), b3, out);
}
